// LinAngularAttention_37314675868447
// MI455X (gfx1250) — matrix operand order repaired and checked by reading, not compiled, not run on hardware
//
#include <hip/hip_runtime.h>

// ---------------------------------------------------------------------------
// LinAngularAttention on MI455X (gfx1250): bf16 WMMA GEMMs + fused attn core
// with Tensor Data Mover (TDM) staging of k/v tiles into LDS.
// ---------------------------------------------------------------------------

typedef __attribute__((ext_vector_type(16))) __bf16 v16bf;
typedef __attribute__((ext_vector_type(8)))  float  v8f;
typedef __attribute__((ext_vector_type(4)))  unsigned int v4u;
typedef __attribute__((ext_vector_type(8)))  int v8i;
typedef __attribute__((ext_vector_type(4)))  int v4i;

#define BDIM   32
#define CDIM   512
#define L_LEN  1024
#define HEADS  8
#define DHEAD  64
#define KSZ    9
#define CHROWS 128         // rows of k/v staged per TDM chunk

__device__ __forceinline__ int lane_id() { return (int)(threadIdx.x & 31); }

union FragU { v16bf v; uint4 q[2]; };

// A fragment 16x32 bf16 from row-major [*, ld] at (row0, k0).
// ISA layout: lanes 0-15 row m, elems K = {0..7, 16..23}; lanes 16-31: {8..15, 24..31}.
__device__ __forceinline__ v16bf load_A_rows(const __bf16* tile, int ld) {
    int l = lane_id(); int half = l >> 4; int m = l & 15;
    const __bf16* p = tile + (size_t)m * ld + half * 8;
    FragU u;
    u.q[0] = *(const uint4*)p;        // K = half*8 + 0..7
    u.q[1] = *(const uint4*)(p + 16); // K = 16 + half*8 + 0..7
    return u.v;
}

// B fragment 32x16 bf16 where the source matrix is stored [N_total, K_total]
// (K contiguous), i.e. B(kk,n) = mat[(n0+n)*ld + k0+kk].
__device__ __forceinline__ v16bf load_B_cols(const __bf16* mat, int ld, int k0, int n0) {
    int l = lane_id(); int half = l >> 4; int n = l & 15;
    const __bf16* p = mat + (size_t)(n0 + n) * ld + k0 + half * 8;
    FragU u;
    u.q[0] = *(const uint4*)p;
    u.q[1] = *(const uint4*)(p + 16);
    return u.v;
}

// A fragment where A(m,kk) = mat[(k0+kk)*ld + m0+m]  (transposed source, LDS)
__device__ __forceinline__ v16bf load_A_transposed(const __bf16* mat, int ld, int k0, int m0) {
    int l = lane_id(); int half = l >> 4; int m = l & 15;
    const __bf16* p = mat + (size_t)k0 * ld + m0 + m;
    v16bf f;
#pragma unroll
    for (int j = 0; j < 8; ++j) f[j]     = p[(size_t)(half * 8 + j) * ld];
#pragma unroll
    for (int j = 0; j < 8; ++j) f[j + 8] = p[(size_t)(16 + half * 8 + j) * ld];
    return f;
}

// B fragment where B(kk,n) = mat[(k0+kk)*ld + n0+n]  (K along rows, LDS)
__device__ __forceinline__ v16bf load_B_strided(const __bf16* mat, int ld, int k0, int n0) {
    int l = lane_id(); int half = l >> 4; int n = l & 15;
    const __bf16* p = mat + (size_t)(k0 + half * 8) * ld + n0 + n;
    v16bf f;
#pragma unroll
    for (int j = 0; j < 16; ++j) f[j] = p[(size_t)(j < 8 ? j : j + 8) * ld];
    return f;
}

__device__ __forceinline__ v8f wmma_bf16(v16bf a, v16bf b, v8f c) {
    return __builtin_amdgcn_wmma_f32_16x16x32_bf16(false, a, false, b, (short)0, c,
                                                   false, false);
}

// ---------------------------------------------------------------------------
// TDM: DMA a [rows x 64] bf16 tile (row-major, contiguous rows of 128B) from
// global into LDS. D# built per CDNA5 ISA ch.8 (2D tensor, groups 2/3 zero).
// This toolchain exposes the 6-arg builtin (extra int32x8 descriptor group).
// ---------------------------------------------------------------------------
__device__ __forceinline__ void tdm_load_tile(const __bf16* gsrc, unsigned lds_off,
                                              unsigned rows, unsigned tensor_rows) {
    unsigned long long ga = (unsigned long long)(uintptr_t)gsrc;
    v4u g0;
    g0[0] = 1u;                                           // count=1, user mode
    g0[1] = lds_off;                                      // lds_addr (bytes)
    g0[2] = (unsigned)(ga & 0xFFFFFFFFu);                 // global_addr[31:0]
    g0[3] = (unsigned)((ga >> 32) & 0x01FFFFFFu)          // global_addr[56:32]
          | (2u << 30);                                   // type = 2 (image)
    v8i g1;
    g1[0] = (int)(1u << 16);                              // data_size=1 (2 bytes)
    g1[1] = (int)((DHEAD & 0xFFFFu) << 16);               // tensor_dim0 = 64
    g1[2] = (int)((tensor_rows & 0xFFFFu) << 16);         // tensor_dim1
    g1[3] = (int)((unsigned)DHEAD << 16);                 // tile_dim0 = 64
    g1[4] = (int)(rows & 0xFFFFu);                        // tile_dim1 = rows
    g1[5] = (int)DHEAD;                                   // tensor_dim0_stride = 64
    g1[6] = 0;
    g1[7] = 0;
    v4i gz4 = {0, 0, 0, 0};
    v8i gz8 = {0, 0, 0, 0, 0, 0, 0, 0};
    __builtin_amdgcn_tensor_load_to_lds(g0, g1, gz4, gz4, gz8, 0);
}

// ---------------------------------------------------------------------------
// K0: pack x [B,C,H,W] -> xf bf16 [B*L, C];  pack weights f32 -> bf16
// ---------------------------------------------------------------------------
__global__ void k_pack_x(const float* __restrict__ x, __bf16* __restrict__ xf) {
    int idx = blockIdx.x * blockDim.x + threadIdx.x;   // over [B, L, C]
    int c = idx & (CDIM - 1);
    int l = (idx >> 9) & (L_LEN - 1);
    int b = idx >> 19;
    xf[idx] = (__bf16)x[((size_t)b * CDIM + c) * L_LEN + l];
}

__global__ void k_pack_w(const float* __restrict__ w, __bf16* __restrict__ wb, int n) {
    int idx = blockIdx.x * blockDim.x + threadIdx.x;
    if (idx < n) wb[idx] = (__bf16)w[idx];
}

// ---------------------------------------------------------------------------
// K1: qkv GEMM  [32768 x 512] * [1536 x 512]^T  -> scatter to q/k/v [B,H,L,d]
// ---------------------------------------------------------------------------
__global__ void k_qkv(const __bf16* __restrict__ xf, const __bf16* __restrict__ wqkv,
                      __bf16* __restrict__ q, __bf16* __restrict__ k,
                      __bf16* __restrict__ v) {
    int wid = blockIdx.x * (blockDim.x >> 5) + (threadIdx.x >> 5);
    int ng = wid % 24;            // 24 N-groups of 64 -> 1536 outputs
    int mi = wid / 24;            // 2048 M-tiles
    int row0 = mi * 16;
    v8f acc[4] = {{}, {}, {}, {}};
    const __bf16* arow = xf + (size_t)row0 * CDIM;
    for (int k0 = 0; k0 < CDIM; k0 += 32) {
        __builtin_prefetch(arow + k0 + 128, 0, 3);  // global_prefetch_b8, near
        v16bf a = load_A_rows(arow + k0, CDIM);
#pragma unroll
        for (int nt = 0; nt < 4; ++nt) {
            v16bf bf = load_B_cols(wqkv, CDIM, k0, ng * 64 + nt * 16);
            acc[nt] = wmma_bf16(a, bf, acc[nt]);
        }
    }
    int b = row0 >> 10;
    int lbase = (row0 & (L_LEN - 1)) + ((lane_id() >> 4) << 3);
    int ncol = lane_id() & 15;
#pragma unroll
    for (int nt = 0; nt < 4; ++nt) {
        int o = ng * 64 + nt * 16 + ncol;
        int which = o >> 9;
        int h = (o >> 6) & 7;
        int e = o & 63;
        __bf16* base = (which == 0) ? q : ((which == 1) ? k : v);
        __bf16* dst = base + ((size_t)(b * HEADS + h) * L_LEN + lbase) * DHEAD + e;
#pragma unroll
        for (int r = 0; r < 8; ++r) dst[(size_t)r * DHEAD] = (__bf16)acc[nt][r];
    }
}

// ---------------------------------------------------------------------------
// K2: L2-normalize q and k rows (along d) in place
// ---------------------------------------------------------------------------
__global__ void k_norm_qk(__bf16* __restrict__ q, __bf16* __restrict__ k) {
    int row = blockIdx.x * blockDim.x + threadIdx.x;   // [0, B*HEADS*L)
    size_t base = (size_t)row * DHEAD;
    float s = 0.f;
    for (int i = 0; i < DHEAD; ++i) { float t = (float)q[base + i]; s += t * t; }
    float inv = rsqrtf(s);
    for (int i = 0; i < DHEAD; ++i) q[base + i] = (__bf16)((float)q[base + i] * inv);
    s = 0.f;
    for (int i = 0; i < DHEAD; ++i) { float t = (float)k[base + i]; s += t * t; }
    inv = rsqrtf(s);
    for (int i = 0; i < DHEAD; ++i) k[base + i] = (__bf16)((float)k[base + i] * inv);
}

// ---------------------------------------------------------------------------
// K2b: depthwise conv over L (9 taps, zero pad) on v
// ---------------------------------------------------------------------------
__global__ void k_dconv(const __bf16* __restrict__ v, const float* __restrict__ wd,
                        __bf16* __restrict__ dc) {
    int idx = blockIdx.x * blockDim.x + threadIdx.x;   // over [B*H, L, d]
    int e = idx & 63;
    int l = (idx >> 6) & (L_LEN - 1);
    int bh = idx >> 16;
    int h = bh & 7;
    const __bf16* vb = v + (size_t)bh * L_LEN * DHEAD;
    float s = 0.f;
#pragma unroll
    for (int j = 0; j < KSZ; ++j) {
        int ls = l + j - KSZ / 2;
        if (ls >= 0 && ls < L_LEN)
            s += wd[h * KSZ + j] * (float)vb[(size_t)ls * DHEAD + e];
    }
    dc[idx] = (__bf16)s;
}

// ---------------------------------------------------------------------------
// K3: fused per-(b,h) core.
// Phase 1: attn = k^T v (64x64, K=1024). k/v chunks (128x64 bf16) staged into
//          LDS by the Tensor Data Mover (tensor_load_to_lds + s_wait_tensorcnt),
//          fragments built from LDS, result stored transposed (bf16) in LDS.
// Phase 2: q @ attn, epilogue 0.5v + (1/pi)*, shfl row-norm, + dconv,
//          store mid bf16 [B*L, C].
// ---------------------------------------------------------------------------
__global__ void k_attn_core(const __bf16* __restrict__ q, const __bf16* __restrict__ k,
                            const __bf16* __restrict__ v, const __bf16* __restrict__ dc,
                            __bf16* __restrict__ mid) {
    __shared__ __bf16 kbuf[CHROWS * DHEAD];   // 16 KB
    __shared__ __bf16 vbuf[CHROWS * DHEAD];   // 16 KB
    __shared__ __bf16 attnT[DHEAD * DHEAD];   // [e2][e1], 8 KB
    int bh = blockIdx.x;
    int b = bh >> 3, h = bh & 7;
    int w = threadIdx.x >> 5;
    size_t base = (size_t)bh * L_LEN * DHEAD;
    const __bf16* kb  = k  + base;
    const __bf16* vb  = v  + base;
    const __bf16* qb  = q  + base;
    const __bf16* dcb = dc + base;

    // ---- phase 1: wave w owns tiles (mi0 = w>>2, ni) and (mi0+2, ni), ni = w&3
    int mi0 = (w >> 2) * 16, mi1 = mi0 + 32, ni = (w & 3) * 16;
    v8f acc0 = {}, acc1 = {};
    for (int ch = 0; ch < L_LEN / CHROWS; ++ch) {
        if (w == 0) {
            tdm_load_tile(kb + (size_t)ch * CHROWS * DHEAD,
                          (unsigned)(uintptr_t)&kbuf[0], CHROWS, L_LEN);
            tdm_load_tile(vb + (size_t)ch * CHROWS * DHEAD,
                          (unsigned)(uintptr_t)&vbuf[0], CHROWS, L_LEN);
            __builtin_amdgcn_s_wait_tensorcnt(0);
        }
        __syncthreads();
#pragma unroll
        for (int l0 = 0; l0 < CHROWS; l0 += 32) {
            v16bf bf = load_B_strided(vbuf, DHEAD, l0, ni);     // B(kk,n)=v[l,n]
            v16bf a0 = load_A_transposed(kbuf, DHEAD, l0, mi0); // A(m,kk)=k[l,m]
            v16bf a1 = load_A_transposed(kbuf, DHEAD, l0, mi1);
            acc0 = wmma_bf16(a0, bf, acc0);
            acc1 = wmma_bf16(a1, bf, acc1);
        }
        __syncthreads();   // all waves done reading before next TDM overwrite
    }
    {
        int lane = lane_id(); int half = lane >> 4; int n = lane & 15;
        int e2 = ni + n;
#pragma unroll
        for (int r = 0; r < 8; ++r) {
            attnT[e2 * DHEAD + (mi0 + half * 8 + r)] = (__bf16)acc0[r];
            attnT[e2 * DHEAD + (mi1 + half * 8 + r)] = (__bf16)acc1[r];
        }
    }
    __syncthreads();

    const float inv_pi = 0.31830988618379067f;
    // ---- phase 2: 64 M-tiles over 8 waves, each wave holds full 16x64 rows ----
    for (int mt = w * 8; mt < w * 8 + 8; ++mt) {
        int l0 = mt * 16;
        v8f acc[4] = {{}, {}, {}, {}};
#pragma unroll
        for (int k0 = 0; k0 < DHEAD; k0 += 32) {
            v16bf a = load_A_rows(qb + (size_t)l0 * DHEAD + k0, DHEAD);
#pragma unroll
            for (int nt = 0; nt < 4; ++nt) {
                v16bf bf = load_B_cols(attnT, DHEAD, k0, nt * 16);
                acc[nt] = wmma_bf16(a, bf, acc[nt]);
            }
        }
        int lane = lane_id(); int half = lane >> 4; int n = lane & 15;
        float outv[4][8];
#pragma unroll
        for (int nt = 0; nt < 4; ++nt) {
            int e = nt * 16 + n;
#pragma unroll
            for (int r = 0; r < 8; ++r) {
                int l = l0 + half * 8 + r;
                outv[nt][r] = 0.5f * (float)vb[(size_t)l * DHEAD + e]
                            + inv_pi * acc[nt][r];
            }
        }
        // row L2 norm: each row's 64 values live in one 16-lane half (4 tiles)
#pragma unroll
        for (int r = 0; r < 8; ++r) {
            float s = 0.f;
#pragma unroll
            for (int nt = 0; nt < 4; ++nt) s += outv[nt][r] * outv[nt][r];
#pragma unroll
            for (int msk = 8; msk >= 1; msk >>= 1) s += __shfl_xor(s, msk, 32);
            float inv = rsqrtf(s);
#pragma unroll
            for (int nt = 0; nt < 4; ++nt) outv[nt][r] *= inv;
        }
#pragma unroll
        for (int nt = 0; nt < 4; ++nt) {
            int e = nt * 16 + n;
#pragma unroll
            for (int r = 0; r < 8; ++r) {
                int l = l0 + half * 8 + r;
                float val = outv[nt][r] + (float)dcb[(size_t)l * DHEAD + e];
                mid[((size_t)(b * L_LEN + l)) * CDIM + h * DHEAD + e] = (__bf16)val;
            }
        }
    }
}

// ---------------------------------------------------------------------------
// K4: output projection  [32768 x 512] * [512 x 512]^T + bias -> [B,C,H,W] f32
// ---------------------------------------------------------------------------
__global__ void k_proj(const __bf16* __restrict__ mid, const __bf16* __restrict__ wproj,
                       const float* __restrict__ bias, float* __restrict__ out) {
    int wid = blockIdx.x * (blockDim.x >> 5) + (threadIdx.x >> 5);
    int ng = wid & 7;           // 8 N-groups of 64 -> 512 outputs
    int mi = wid >> 3;          // 2048 M-tiles
    int row0 = mi * 16;
    v8f acc[4] = {{}, {}, {}, {}};
    const __bf16* arow = mid + (size_t)row0 * CDIM;
    for (int k0 = 0; k0 < CDIM; k0 += 32) {
        __builtin_prefetch(arow + k0 + 128, 0, 3);
        v16bf a = load_A_rows(arow + k0, CDIM);
#pragma unroll
        for (int nt = 0; nt < 4; ++nt) {
            v16bf bf = load_B_cols(wproj, CDIM, k0, ng * 64 + nt * 16);
            acc[nt] = wmma_bf16(a, bf, acc[nt]);
        }
    }
    int lane = lane_id(); int half = lane >> 4; int n = lane & 15;
    int bidx = row0 >> 10;
    int l0 = (row0 & (L_LEN - 1)) + half * 8;
#pragma unroll
    for (int nt = 0; nt < 4; ++nt) {
        int o = ng * 64 + nt * 16 + n;
        float bv = bias[o];
        float* dst = out + ((size_t)(bidx * CDIM + o)) * L_LEN + l0;
        float4 s0 = make_float4(acc[nt][0] + bv, acc[nt][1] + bv,
                                acc[nt][2] + bv, acc[nt][3] + bv);
        float4 s1 = make_float4(acc[nt][4] + bv, acc[nt][5] + bv,
                                acc[nt][6] + bv, acc[nt][7] + bv);
        *(float4*)dst = s0;
        *(float4*)(dst + 4) = s1;
    }
}

// ---------------------------------------------------------------------------
extern "C" void kernel_launch(void* const* d_in, const int* in_sizes, int n_in,
                              void* d_out, int out_size, void* d_ws, size_t ws_size,
                              hipStream_t stream) {
    const float* x       = (const float*)d_in[0];
    const float* w_qkv   = (const float*)d_in[1];
    const float* w_proj  = (const float*)d_in[2];
    const float* b_proj  = (const float*)d_in[3];
    const float* w_dconv = (const float*)d_in[4];
    float* out = (float*)d_out;

    char* ws = (char*)d_ws;
    size_t off = 0;
    auto alloc = [&](size_t bytes) -> char* {
        char* p = ws + off;
        off += (bytes + 255) & ~(size_t)255;
        return p;
    };
    const size_t n_bhld = (size_t)BDIM * HEADS * L_LEN * DHEAD;       // 16.78M elems
    __bf16* xf     = (__bf16*)alloc((size_t)BDIM * L_LEN * CDIM * 2); // also "mid"
    __bf16* wqkvb  = (__bf16*)alloc((size_t)3 * CDIM * CDIM * 2);
    __bf16* wprojb = (__bf16*)alloc((size_t)CDIM * CDIM * 2);
    __bf16* qb     = (__bf16*)alloc(n_bhld * 2);
    __bf16* kb     = (__bf16*)alloc(n_bhld * 2);
    __bf16* vb     = (__bf16*)alloc(n_bhld * 2);
    __bf16* dcb    = (__bf16*)alloc(n_bhld * 2);
    __bf16* mid    = xf;   // xf dead after k_qkv; reuse for mid

    k_pack_x<<<65536, 256, 0, stream>>>(x, xf);
    k_pack_w<<<3072, 256, 0, stream>>>(w_qkv, wqkvb, 3 * CDIM * CDIM);
    k_pack_w<<<1024, 256, 0, stream>>>(w_proj, wprojb, CDIM * CDIM);
    k_qkv<<<6144, 256, 0, stream>>>(xf, wqkvb, qb, kb, vb);      // 49152 waves
    k_norm_qk<<<1024, 256, 0, stream>>>(qb, kb);
    k_dconv<<<65536, 256, 0, stream>>>(vb, w_dconv, dcb);
    k_attn_core<<<BDIM * HEADS, 256, 0, stream>>>(qb, kb, vb, dcb, mid);
    k_proj<<<2048, 256, 0, stream>>>(mid, wprojb, b_proj, out);  // 16384 waves
}
